// GraphFeatureTokenizer_82660940579489
// MI455X (gfx1250) — compile-verified, matrix operand order checked by reading, not run on hardware
//
#include <hip/hip_runtime.h>
#include <hip/hip_bf16.h>

// ---------- problem constants (mirror the reference module constants) ----------
#define DN    128   // node/edge feature dim
#define DP    64    // node identifier dim
#define HD    768   // hidden dim
#define KTOT  256   // effective K after folding E_V/E_E into the bias
#define NPG   512   // max_node_num
#define EPG   4096  // max_edge_num
#define SEQ   (1 + NPG + EPG)   // 4609

// ---------- vector types ----------
typedef __attribute__((ext_vector_type(16))) __bf16 v16bf;
typedef __attribute__((ext_vector_type(8)))  __bf16 v8bf;
typedef __attribute__((ext_vector_type(4)))  __bf16 v4bf;
typedef __attribute__((ext_vector_type(8)))  float  v8f;

__device__ __forceinline__ __bf16 f2bf(float f) {
    unsigned u = __builtin_bit_cast(unsigned, f);
    u = (u + 0x7FFFu + ((u >> 16) & 1u)) >> 16;      // round-to-nearest-even
    unsigned short s = (unsigned short)u;
    return __builtin_bit_cast(__bf16, s);
}

// ---------- kernel 0: zero the per-graph counters ----------
__global__ void gft_zero_kernel(int* cnt, int n) {
    int i = blockIdx.x * blockDim.x + threadIdx.x;
    if (i < n) cnt[i] = 0;
}

// ---------- kernel 1: per-graph node / edge counts ----------
__global__ void gft_count_kernel(const int* __restrict__ batch,
                                 const int* __restrict__ srcIdx,
                                 int Nn, int E,
                                 int* __restrict__ ncnt, int* __restrict__ ecnt) {
    int i = blockIdx.x * blockDim.x + threadIdx.x;
    if (i < Nn) {
        atomicAdd(&ncnt[batch[i]], 1);
    } else if (i < Nn + E) {
        int j = i - Nn;
        atomicAdd(&ecnt[batch[srcIdx[j]]], 1);
    }
}

// ---------- kernel 2: exclusive scan (G is tiny) ----------
__global__ void gft_scan_kernel(const int* __restrict__ ncnt, const int* __restrict__ ecnt,
                                int* __restrict__ nptr, int* __restrict__ eptr, int G) {
    if (threadIdx.x == 0 && blockIdx.x == 0) {
        int rn = 0, re = 0;
        for (int g = 0; g < G; ++g) {
            nptr[g] = rn; rn += ncnt[g];
            eptr[g] = re; re += ecnt[g];
        }
    }
}

// ---------- kernel 3: convert W[:, :256] to bf16; fold E_V/E_E + b into c_v/c_e ----------
__global__ void gft_prep_kernel(const float* __restrict__ W, const float* __restrict__ b,
                                const float* __restrict__ EV, const float* __restrict__ EE,
                                __bf16* __restrict__ Wb,
                                float* __restrict__ cv, float* __restrict__ ce) {
    int h = blockIdx.x;          // 0..HD-1
    int t = threadIdx.x;         // 0..63
    const float* wr = W + (size_t)h * (KTOT + DP);      // row stride = IN_DIM = 320
    // bf16 copy of the first 256 columns
    for (int c = t; c < KTOT; c += 64) Wb[(size_t)h * KTOT + c] = f2bf(wr[c]);
    // dot( W[h, 256:320], E_id )
    __shared__ float sv[64], se[64];
    float wk = wr[KTOT + t];
    sv[t] = wk * EV[t];
    se[t] = wk * EE[t];
    __syncthreads();
    for (int o = 32; o > 0; o >>= 1) {
        if (t < o) { sv[t] += sv[t + o]; se[t] += se[t + o]; }
        __syncthreads();
    }
    if (t == 0) { cv[h] = b[h] + sv[0]; ce[h] = b[h] + se[0]; }
}

// ---------- kernel 4: fused node+edge WMMA GEMM with scattered epilogue ----------
// C[r, h] = sum_k A[r,k] * Wb[h,k] + c_type[h], scattered into tokens[g, slot, h]
#define BM 64
#define BN 128
#define LDA 264   // bf16 elements per LDS row (264*2 B, keeps 16-B alignment, skews banks)

__global__ __launch_bounds__(256)
void gft_gemm_kernel(const float* __restrict__ x, const float* __restrict__ ea,
                     const float* __restrict__ P,
                     const int* __restrict__ srcIdx, const int* __restrict__ dstIdx,
                     const int* __restrict__ batch,
                     const __bf16* __restrict__ Wb,
                     const float* __restrict__ cv, const float* __restrict__ ce,
                     const int* __restrict__ nptr, const int* __restrict__ eptr,
                     float* __restrict__ out, int Nn, int E) {
    __shared__ __bf16 At[BM * LDA];

    const int Mtot = Nn + E;
    const int row0 = blockIdx.y * BM;
    const int h0   = blockIdx.x * BN;
    const int tid  = threadIdx.x;

    // ---- assemble A tile [BM x 256] in LDS as bf16 (gather + convert) ----
    // 64 float4 per row, BM*64 = 4096 float4 over 256 threads
    for (int i = tid; i < BM * 64; i += 256) {
        int rl  = i >> 6;
        int col = (i & 63) << 2;        // 0..252, step 4
        int r   = row0 + rl;
        float4 v = make_float4(0.f, 0.f, 0.f, 0.f);
        if (r < Mtot) {
            const float* src;
            if (r < Nn) {                                  // node row: [x | P | P]
                if (col < DN)            src = x + (size_t)r * DN + col;
                else if (col < DN + DP)  src = P + (size_t)r * DP + (col - DN);
                else                     src = P + (size_t)r * DP + (col - DN - DP);
            } else {                                       // edge row: [ea | P_u | P_v]
                int j = r - Nn;
                if (col < DN)            src = ea + (size_t)j * DN + col;
                else if (col < DN + DP)  src = P + (size_t)srcIdx[j] * DP + (col - DN);
                else                     src = P + (size_t)dstIdx[j] * DP + (col - DN - DP);
            }
            v = *(const float4*)src;
        }
        v4bf w = { f2bf(v.x), f2bf(v.y), f2bf(v.z), f2bf(v.w) };
        *(v4bf*)&At[rl * LDA + col] = w;
    }
    __syncthreads();

    // ---- WMMA: 8 waves, each owns a 32x32 patch (2x2 tiles of 16x16) ----
    const int wave = tid >> 5;
    const int lane = tid & 31;
    const int wm   = wave >> 2;          // 0..1  -> rows  wm*32
    const int wn   = wave & 3;           // 0..3  -> cols  wn*32
    const int lrow  = lane & 15;
    const int khalf = (lane >> 4) << 3;  // 0 or 8 (ISA 16-bit fragment layout)

    v8f acc[2][2] = {};

    for (int kk = 0; kk < KTOT; kk += 32) {
        v16bf afrag[2], bfrag[2];
        #pragma unroll
        for (int ms = 0; ms < 2; ++ms) {
            const __bf16* ap = &At[(wm * 32 + ms * 16 + lrow) * LDA + kk + khalf];
            v8bf lo = *(const v8bf*)ap;
            v8bf hi = *(const v8bf*)(ap + 16);
            afrag[ms] = __builtin_shufflevector(lo, hi, 0,1,2,3,4,5,6,7,8,9,10,11,12,13,14,15);
        }
        #pragma unroll
        for (int ns = 0; ns < 2; ++ns) {
            int h = h0 + wn * 32 + ns * 16 + lrow;
            const __bf16* bp = Wb + (size_t)h * KTOT + kk + khalf;
            v8bf lo = *(const v8bf*)bp;
            v8bf hi = *(const v8bf*)(bp + 16);
            bfrag[ns] = __builtin_shufflevector(lo, hi, 0,1,2,3,4,5,6,7,8,9,10,11,12,13,14,15);
        }
        #pragma unroll
        for (int ms = 0; ms < 2; ++ms)
            #pragma unroll
            for (int ns = 0; ns < 2; ++ns)
                acc[ms][ns] = __builtin_amdgcn_wmma_f32_16x16x32_bf16(
                    false, afrag[ms], false, bfrag[ns],
                    (short)0, acc[ms][ns], false, false);
    }

    // ---- epilogue: add folded bias, scatter to tokens[g, slot, h] ----
    const int mhi = (lane >> 4) << 3;    // C layout: lanes 16-31 hold M = v + 8
    #pragma unroll
    for (int ms = 0; ms < 2; ++ms) {
        #pragma unroll
        for (int ns = 0; ns < 2; ++ns) {
            int nn = h0 + wn * 32 + ns * 16 + (lane & 15);
            #pragma unroll
            for (int v = 0; v < 8; ++v) {
                int r = row0 + wm * 32 + ms * 16 + v + mhi;
                if (r >= Mtot) continue;
                int g, orow; float cadd;
                if (r < Nn) {
                    g    = batch[r];
                    orow = g * SEQ + 1 + (r - nptr[g]);
                    cadd = cv[nn];
                } else {
                    int j = r - Nn;
                    g    = batch[srcIdx[j]];
                    orow = g * SEQ + 1 + NPG + (j - eptr[g]);
                    cadd = ce[nn];
                }
                out[(size_t)orow * HD + nn] = acc[ms][ns][v] + cadd;
            }
        }
    }
}

// ---------- kernel 5: graph token, masks, zero padded slots ----------
__global__ void gft_fill_kernel(const float* __restrict__ gtok,
                                const int* __restrict__ ncnt, const int* __restrict__ ecnt,
                                float* __restrict__ out, float* __restrict__ mask) {
    int g = blockIdx.x;
    int t = threadIdx.x;
    float* base = out + (size_t)g * SEQ * HD;
    for (int i = t; i < HD; i += 256) base[i] = gtok[i];     // tokens[g, 0, :]
    int nc = ncnt[g], ec = ecnt[g];
    for (int s = t; s < SEQ; s += 256) {
        bool valid;
        if (s == 0)            valid = true;
        else if (s <= NPG)     valid = (s - 1) < nc;
        else                   valid = (s - 1 - NPG) < ec;
        mask[(size_t)g * SEQ + s] = valid ? 1.0f : 0.0f;
        if (!valid) {
            float* rowp = base + (size_t)s * HD;
            for (int i = 0; i < HD; ++i) rowp[i] = 0.0f;
        }
    }
}

extern "C" void kernel_launch(void* const* d_in, const int* in_sizes, int n_in,
                              void* d_out, int out_size, void* d_ws, size_t ws_size,
                              hipStream_t stream) {
    const float* x    = (const float*)d_in[0];
    const float* ea   = (const float*)d_in[1];
    const float* P    = (const float*)d_in[2];
    const float* EV   = (const float*)d_in[3];
    const float* EE   = (const float*)d_in[4];
    const float* W    = (const float*)d_in[5];
    const float* b    = (const float*)d_in[6];
    const float* gtok = (const float*)d_in[7];
    const int*   ei   = (const int*)d_in[8];
    const int*   bat  = (const int*)d_in[9];

    const int Nn = in_sizes[0] / DN;        // 16384
    const int E  = in_sizes[1] / DN;        // 131072
    const int G  = Nn / NPG;                // 32
    const int*   src = ei;
    const int*   dst = ei + E;

    // workspace layout
    char*   ws   = (char*)d_ws;
    __bf16* Wb   = (__bf16*)ws;                              // HD*KTOT*2 = 393216 B
    float*  cv   = (float*)(ws + (size_t)HD * KTOT * 2);     // 768 f
    float*  ce   = cv + HD;                                  // 768 f
    int*    ncnt = (int*)(ce + HD);
    int*    ecnt = ncnt + G;
    int*    nptr = ecnt + G;
    int*    eptr = nptr + G;

    float* out  = (float*)d_out;
    float* mask = out + (size_t)G * SEQ * HD;

    gft_zero_kernel<<<1, 256, 0, stream>>>(ncnt, 2 * G);
    gft_count_kernel<<<(Nn + E + 255) / 256, 256, 0, stream>>>(bat, src, Nn, E, ncnt, ecnt);
    gft_scan_kernel<<<1, 32, 0, stream>>>(ncnt, ecnt, nptr, eptr, G);
    gft_prep_kernel<<<HD, 64, 0, stream>>>(W, b, EV, EE, Wb, cv, ce);

    dim3 grid(HD / BN, (Nn + E + BM - 1) / BM);              // (6, 2304)
    gft_gemm_kernel<<<grid, 256, 0, stream>>>(x, ea, P, src, dst, bat,
                                              Wb, cv, ce, nptr, eptr, out, Nn, E);
    gft_fill_kernel<<<G, 256, 0, stream>>>(gtok, ncnt, ecnt, out, mask);
}
